// YuanMoeLayer_3332894622515
// MI455X (gfx1250) — matrix operand order repaired and checked by reading, not compiled, not run on hardware
//
#include <hip/hip_runtime.h>
#include <math.h>

typedef __attribute__((ext_vector_type(16))) __bf16 v16bf;
typedef __attribute__((ext_vector_type(8)))  float  v8f;

#define H_DIM 1024
#define F_DIM 2816
#define NEXP  8
#define NTOK  4096
#define CAP   8192          // bucket capacity per expert (N*K worst case)
#define MROWS 32            // token rows per workgroup tile
#define TILES_PER_E (CAP / MROWS)   // 256
#define XSTR  1040          // padded X-tile LDS stride (elems), breaks bank conflicts
#define ASTR  2832          // padded act LDS stride (elems)

// ---------------- fp32 -> bf16 conversion (4 elems / thread) ----------------
__global__ void cvt_f32_bf16_x4(const float4* __restrict__ src, uint2* __restrict__ dst, int n4) {
    int i = blockIdx.x * blockDim.x + threadIdx.x;
    int stride = gridDim.x * blockDim.x;
    for (; i < n4; i += stride) {
        float4 f = src[i];
        union { __bf16 h[4]; uint2 u; } p;
        p.h[0] = (__bf16)f.x; p.h[1] = (__bf16)f.y;
        p.h[2] = (__bf16)f.z; p.h[3] = (__bf16)f.w;
        dst[i] = p.u;
    }
}

__global__ void zero_counts(int* counts) {
    if (threadIdx.x < NEXP) counts[threadIdx.x] = 0;
}

// ---------------- router: mixed -> attn -> top2 -> scatter ----------------
__global__ void router_kernel(const float* __restrict__ x, const float* __restrict__ wqkv,
                              float* __restrict__ scoreArr, int* __restrict__ counts,
                              int* __restrict__ bucket) {
    __shared__ float mixedLds[8][24];
    __shared__ float logitLds[8][8];
    int wave = threadIdx.x >> 5, lane = threadIdx.x & 31;
    int n = blockIdx.x * 8 + wave;
    const float* xr = x + (long)n * H_DIM;
    for (int o = 0; o < 24; ++o) {
        float s = 0.f;
        const float* wr = wqkv + o * H_DIM;
        for (int h = lane; h < H_DIM; h += 32) s += xr[h] * wr[h];
        for (int off = 16; off; off >>= 1) s += __shfl_xor(s, off);
        if (lane == 0) mixedLds[wave][o] = s;
    }
    __syncthreads();
    if (lane < 8) {   // lane e computes logit[e]
        float q = mixedLds[wave][lane];
        float m[8], mx = -1e30f;
        #pragma unroll
        for (int j = 0; j < 8; ++j) { m[j] = q * mixedLds[wave][8 + j]; mx = fmaxf(mx, m[j]); }
        float sum = 0.f, acc = 0.f;
        #pragma unroll
        for (int j = 0; j < 8; ++j) {
            float p = __expf(m[j] - mx);
            sum += p; acc += p * mixedLds[wave][16 + j];
        }
        logitLds[wave][lane] = acc / sum;
    }
    __syncthreads();
    if (lane == 0) {
        float l0 = -1e30f; int i0 = 0;
        for (int j = 0; j < 8; ++j) { float lj = logitLds[wave][j]; if (lj > l0) { l0 = lj; i0 = j; } }
        float l1 = -1e30f; int i1 = 0;
        for (int j = 0; j < 8; ++j) { if (j == i0) continue; float lj = logitLds[wave][j]; if (lj > l1) { l1 = lj; i1 = j; } }
        float e1 = __expf(l1 - l0);
        float inv = 1.f / (1.f + e1);
        scoreArr[2 * n]     = inv;
        scoreArr[2 * n + 1] = e1 * inv;
        int p0 = atomicAdd(&counts[i0], 1); bucket[i0 * CAP + p0] = 2 * n;
        int p1 = atomicAdd(&counts[i1], 1); bucket[i1 * CAP + p1] = 2 * n + 1;
    }
}

// ---------------- fused expert FFN: x @ w1[e].T -> GLU -> @ w2[e].T ----------------
// One workgroup = (expert e, tile of 32 token-slots). 8 waves.
// Phase 1: act[32 x 2816] bf16 in LDS.  Phase 2: out[32 x 1024] -> ypairs.
// Each streamed B-fragment feeds 2 WMMAs (two M-halves) -> halves weight traffic.
__global__ void __launch_bounds__(256) moe_ffn_kernel(
        const __bf16* __restrict__ xbf, const __bf16* __restrict__ w1bf,
        const __bf16* __restrict__ w2bf, const int* __restrict__ counts,
        const int* __restrict__ bucket, float* __restrict__ ypairs) {
    extern __shared__ char smemRaw[];
    int*    slotsLds = (int*)smemRaw;                                  // 32 ints (128B)
    __bf16* xt  = (__bf16*)(smemRaw + 128);                            // 32 x XSTR
    __bf16* act = (__bf16*)(smemRaw + 128 + MROWS * XSTR * 2);         // 32 x ASTR

    int e    = blockIdx.x >> 8;
    int tile = blockIdx.x & (TILES_PER_E - 1);
    int cnt  = counts[e];
    int rowBase = tile * MROWS;
    if (rowBase >= cnt) return;
    int nvalid = cnt - rowBase; if (nvalid > MROWS) nvalid = MROWS;

    int tid = threadIdx.x;
    if (tid < MROWS) slotsLds[tid] = (tid < nvalid) ? bucket[e * CAP + rowBase + tid] : -1;
    __syncthreads();

    // stage X tile (32 tokens x 1024 bf16) into LDS; zeros for invalid rows.
    // 8 threads per row, 16 x 16B chunks each.
    {
        int r = tid >> 3, c = tid & 7;
        int slot = slotsLds[r];
        if (slot >= 0) {
            const __bf16* src = xbf + (long)(slot >> 1) * H_DIM;
            #pragma unroll
            for (int i = 0; i < 16; ++i) {
                int off = (c + i * 8) * 8;                             // 8 bf16 per 16B chunk
                *(uint4*)(xt + r * XSTR + off) = *(const uint4*)(src + off);
            }
        } else {
            #pragma unroll
            for (int i = 0; i < 16; ++i) {
                int off = (c + i * 8) * 8;
                *(uint4*)(xt + r * XSTR + off) = make_uint4(0, 0, 0, 0);
            }
        }
    }
    __syncthreads();

    int wave = tid >> 5, lane = tid & 31;
    int laneRow = lane & 15;
    int hiHalf  = lane >> 4;

    // ---- phase 1: act = silu(x@W1a.T) * (x@W1g.T) ----
    const __bf16* w1e = w1bf + (long)e * (2 * F_DIM) * H_DIM;
    for (int t = wave; t < F_DIM / 16; t += 8) {                       // 176 N-tiles
        v8f accA0 = {0,0,0,0,0,0,0,0}, accG0 = {0,0,0,0,0,0,0,0};
        v8f accA1 = {0,0,0,0,0,0,0,0}, accG1 = {0,0,0,0,0,0,0,0};
        const __bf16* bA = w1e + (long)(t * 16 + laneRow) * H_DIM;              // a-col
        const __bf16* bG = w1e + (long)(F_DIM + t * 16 + laneRow) * H_DIM;      // g-col
        for (int kb = 0; kb < H_DIM; kb += 32) {
            union { v16bf v; uint4 u[2]; } af0, af1;
            const __bf16* ap0 = xt + laneRow * XSTR + kb + hiHalf * 8;
            af0.u[0] = *(const uint4*)ap0;
            af0.u[1] = *(const uint4*)(ap0 + 16);
            const __bf16* ap1 = ap0 + 16 * XSTR;
            af1.u[0] = *(const uint4*)ap1;
            af1.u[1] = *(const uint4*)(ap1 + 16);
            union { v16bf v; uint4 u[2]; } fa, fg;
            const __bf16* pA = bA + kb + hiHalf * 8;
            fa.u[0] = *(const uint4*)pA; fa.u[1] = *(const uint4*)(pA + 16);
            const __bf16* pG = bG + kb + hiHalf * 8;
            fg.u[0] = *(const uint4*)pG; fg.u[1] = *(const uint4*)(pG + 16);
            accA0 = __builtin_amdgcn_wmma_f32_16x16x32_bf16(false, af0.v, false, fa.v,
                                                            (short)0, accA0, false, false);
            accA1 = __builtin_amdgcn_wmma_f32_16x16x32_bf16(false, af1.v, false, fa.v,
                                                            (short)0, accA1, false, false);
            accG0 = __builtin_amdgcn_wmma_f32_16x16x32_bf16(false, af0.v, false, fg.v,
                                                            (short)0, accG0, false, false);
            accG1 = __builtin_amdgcn_wmma_f32_16x16x32_bf16(false, af1.v, false, fg.v,
                                                            (short)0, accG1, false, false);
        }
        #pragma unroll
        for (int i = 0; i < 8; ++i) {                                  // GLU epilogue
            int r = hiHalf ? (i + 8) : i;
            float a0 = accA0[i], g0 = accG0[i];
            act[r * ASTR + t * 16 + laneRow] = (__bf16)((a0 / (1.f + __expf(-a0))) * g0);
            float a1 = accA1[i], g1 = accG1[i];
            act[(r + 16) * ASTR + t * 16 + laneRow] = (__bf16)((a1 / (1.f + __expf(-a1))) * g1);
        }
    }
    __syncthreads();

    // ---- phase 2: out[32 x 1024] = act @ w2[e].T ; wave owns 128 cols ----
    const __bf16* w2e = w2bf + (long)e * H_DIM * F_DIM;
    v8f acc0[8], acc1[8];
    #pragma unroll
    for (int j = 0; j < 8; ++j) { acc0[j] = (v8f){0,0,0,0,0,0,0,0}; acc1[j] = (v8f){0,0,0,0,0,0,0,0}; }
    int colBase0 = wave * 128;
    for (int kb = 0; kb < F_DIM; kb += 32) {                           // 88 K-steps
        union { v16bf v; uint4 u[2]; } af0, af1;
        const __bf16* ap0 = act + laneRow * ASTR + kb + hiHalf * 8;
        af0.u[0] = *(const uint4*)ap0;
        af0.u[1] = *(const uint4*)(ap0 + 16);
        const __bf16* ap1 = ap0 + 16 * ASTR;
        af1.u[0] = *(const uint4*)ap1;
        af1.u[1] = *(const uint4*)(ap1 + 16);
        #pragma unroll
        for (int j = 0; j < 8; ++j) {
            union { v16bf v; uint4 u[2]; } bfrag;
            const __bf16* p = w2e + (long)(colBase0 + j * 16 + laneRow) * F_DIM + kb + hiHalf * 8;
            bfrag.u[0] = *(const uint4*)p; bfrag.u[1] = *(const uint4*)(p + 16);
            acc0[j] = __builtin_amdgcn_wmma_f32_16x16x32_bf16(false, af0.v, false, bfrag.v,
                                                              (short)0, acc0[j], false, false);
            acc1[j] = __builtin_amdgcn_wmma_f32_16x16x32_bf16(false, af1.v, false, bfrag.v,
                                                              (short)0, acc1[j], false, false);
        }
    }
    #pragma unroll
    for (int j = 0; j < 8; ++j) {
        #pragma unroll
        for (int i = 0; i < 8; ++i) {
            int r = hiHalf ? (i + 8) : i;
            int col = colBase0 + j * 16 + laneRow;
            int slot0 = slotsLds[r];
            if (slot0 >= 0) ypairs[(long)slot0 * H_DIM + col] = acc0[j][i];
            int slot1 = slotsLds[r + 16];
            if (slot1 >= 0) ypairs[(long)slot1 * H_DIM + col] = acc1[j][i];
        }
    }
}

// ---------------- combine: out[n] = s0*y[2n] + s1*y[2n+1] ----------------
__global__ void combine_kernel(const float* __restrict__ ypairs, const float* __restrict__ scoreArr,
                               float* __restrict__ out, int total) {
    int i = blockIdx.x * blockDim.x + threadIdx.x;
    if (i >= total) return;
    int n = i >> 10;                 // H_DIM = 1024
    int h = i & 1023;
    out[i] = scoreArr[2 * n]     * ypairs[(long)(2 * n)     * H_DIM + h]
           + scoreArr[2 * n + 1] * ypairs[(long)(2 * n + 1) * H_DIM + h];
}

extern "C" void kernel_launch(void* const* d_in, const int* in_sizes, int n_in,
                              void* d_out, int out_size, void* d_ws, size_t ws_size,
                              hipStream_t stream) {
    const float* x    = (const float*)d_in[0];   // (2,2048,1024) fp32
    const float* wqkv = (const float*)d_in[1];   // (24,1024)
    const float* w1   = (const float*)d_in[2];   // (8,5632,1024)
    const float* w2   = (const float*)d_in[3];   // (8,1024,2816)
    float* out = (float*)d_out;

    char* ws = (char*)d_ws;
    size_t off = 0;
    auto alloc = [&](size_t bytes) -> char* {
        char* p = ws + off;
        off = (off + bytes + 255) & ~(size_t)255;
        return p;
    };
    int*    counts   = (int*)   alloc(NEXP * sizeof(int));
    int*    bucket   = (int*)   alloc((size_t)NEXP * CAP * sizeof(int));
    float*  scoreArr = (float*) alloc((size_t)2 * NTOK * sizeof(float));
    float*  ypairs   = (float*) alloc((size_t)2 * NTOK * H_DIM * sizeof(float));
    __bf16* xbf      = (__bf16*)alloc((size_t)NTOK * H_DIM * 2);
    __bf16* w1bf     = (__bf16*)alloc((size_t)NEXP * 2 * F_DIM * H_DIM * 2);
    __bf16* w2bf     = (__bf16*)alloc((size_t)NEXP * H_DIM * F_DIM * 2);

    zero_counts<<<1, 32, 0, stream>>>(counts);
    cvt_f32_bf16_x4<<<1024, 256, 0, stream>>>((const float4*)x,  (uint2*)xbf,  NTOK * H_DIM / 4);
    cvt_f32_bf16_x4<<<4096, 256, 0, stream>>>((const float4*)w1, (uint2*)w1bf, NEXP * 2 * F_DIM * H_DIM / 4);
    cvt_f32_bf16_x4<<<4096, 256, 0, stream>>>((const float4*)w2, (uint2*)w2bf, NEXP * H_DIM * F_DIM / 4);
    router_kernel<<<NTOK / 8, 256, 0, stream>>>(x, wqkv, scoreArr, counts, bucket);

    size_t ldsBytes = 128 + (size_t)MROWS * XSTR * 2 + (size_t)MROWS * ASTR * 2;  // ~242 KB
    moe_ffn_kernel<<<NEXP * TILES_PER_E, 256, ldsBytes, stream>>>(xbf, w1bf, w2bf, counts, bucket, ypairs);

    combine_kernel<<<(NTOK * H_DIM + 255) / 256, 256, 0, stream>>>(ypairs, scoreArr, out, NTOK * H_DIM);
}
